// GCMC_28054726377858
// MI455X (gfx1250) — compile-verified, matrix operand order checked by reading, not run on hardware
//
#include <hip/hip_runtime.h>
#include <cstddef>

// ---------------------------------------------------------------------------
// Problem constants (from reference)
// ---------------------------------------------------------------------------
static constexpr int kU = 2000, kV = 2000, kF = 2000, kR = 5, kH = 64;
static constexpr int kO = 75, kSH = 64, kSF = 128;
static constexpr int kKF  = 2016;  // padded F for bf16 Bt (Wt) row length
static constexpr int kKUV = 2016;  // padded U/V for bf16 Bt (P^T) row length
static constexpr int kNW  = 384;   // padded R*H (=320) Bt rows
static constexpr int kCAT = 704;   // 2*R*H + SH (multiple of 32)
static constexpr int kKO  = 96;    // padded O (=75) used as K
static constexpr int kNQ  = 128;   // padded O used as Bt rows
static constexpr int kNGV = 2048;  // padded V used as Bt rows for score GEMM
static constexpr int kLDO = 80;    // ld of f32 pre-BN cat output

// ---------------------------------------------------------------------------
// Types for WMMA
// ---------------------------------------------------------------------------
typedef __bf16 bf16_t;
typedef __attribute__((ext_vector_type(16))) __bf16        bf16x16;
typedef __attribute__((ext_vector_type(8)))  float         f32x8;
typedef __attribute__((ext_vector_type(4)))  float         f32x4;
typedef __attribute__((ext_vector_type(4)))  unsigned int  u32x4;

union FragU { u32x4 q[2]; bf16x16 v; bf16_t e[16]; };

// bf16 A/B fragment for v_wmma_*_16x16x32_bf16:
// lane&15 = row, (lane>>4)*8 = K sub-offset; two contiguous 16B chunks
// (K = kb..kb+7 in VGPR0..3, K = kb+16..kb+23 in VGPR4..7).
__device__ __forceinline__ bf16x16 load_frag_bf16(const bf16_t* p) {
  FragU f;
  f.q[0] = *(const u32x4*)(p);
  f.q[1] = *(const u32x4*)(p + 16);
  return f.v;
}

// Fused-cast fragment: read 8 consecutive f32, convert to bf16 in-register.
__device__ __forceinline__ void cvt8(const float* p, bf16_t* e) {
  f32x4 f0 = *(const f32x4*)(p);
  f32x4 f1 = *(const f32x4*)(p + 4);
#pragma unroll
  for (int i = 0; i < 4; ++i) { e[i] = (bf16_t)f0[i]; e[4 + i] = (bf16_t)f1[i]; }
}

template <bool HI_OK>
__device__ __forceinline__ bf16x16 load_frag_f32(const float* p) {
  FragU f;
  cvt8(p, f.e);
  if constexpr (HI_OK) {
    cvt8(p + 16, f.e + 8);
  } else {
#pragma unroll
    for (int i = 8; i < 16; ++i) f.e[i] = (bf16_t)0.0f;  // zero K tail
  }
  return f.v;
}

__device__ __forceinline__ f32x8 wmma_bf16(bf16x16 a, bf16x16 b, f32x8 c) {
  return __builtin_amdgcn_wmma_f32_16x16x32_bf16(false, a, false, b,
                                                 (short)0, c, false, false);
}

// ---------------------------------------------------------------------------
// Generic NT GEMM: C[M,Nreal] = A[M,Kreal] * Bt[N,Kreal]^T  (fp32 accum)
//   WN = waves along N (block tile = (8/WN)*32 rows x WN*32 cols)
//   AF32 = true: A is f32 (cast to bf16 in-register, streamed once from HBM)
//   Bt is always bf16 with row stride ldb (zero padded cols/rows).
// Epilogue: optional bias, relu, bf16 store @ column offset, bf16 transposed
// store, f32 store.
// ---------------------------------------------------------------------------
template <int WN, bool AF32>
__global__ __launch_bounds__(256)
void gemm_nt(const void* __restrict__ Av, const bf16_t* __restrict__ Bt,
             int M, int Kreal, int lda, int ldb,
             bf16_t* Cb, int ldcb, int coffb,
             bf16_t* Ct, int ldct,
             float*  Cf, int ldcf,
             const float* bias, int Nreal, int relu)
{
  const int lane = threadIdx.x & 31;
  const int wave = threadIdx.x >> 5;
  const int m0 = blockIdx.y * ((8 / WN) * 32) + (wave / WN) * 32;
  const int n0 = blockIdx.x * (WN * 32)       + (wave % WN) * 32;
  const int r16 = lane & 15;
  const int kb  = (lane >> 4) << 3;  // 0 or 8

  // Clamp A rows into the real range (pad tiles give dead results; stores are
  // bounds-checked).  Bt buffers are allocated with padded rows.
  int am0 = m0 + r16;      if (am0 >= M) am0 = M - 1;
  int am1 = m0 + 16 + r16; if (am1 >= M) am1 = M - 1;

  const float*  a0f = nullptr, *a1f = nullptr;
  const bf16_t* a0h = nullptr, *a1h = nullptr;
  if constexpr (AF32) {
    a0f = (const float*)Av + (size_t)am0 * lda + kb;
    a1f = (const float*)Av + (size_t)am1 * lda + kb;
  } else {
    a0h = (const bf16_t*)Av + (size_t)am0 * lda + kb;
    a1h = (const bf16_t*)Av + (size_t)am1 * lda + kb;
  }
  const bf16_t* b0p = Bt + (size_t)(n0 + r16)      * ldb + kb;
  const bf16_t* b1p = Bt + (size_t)(n0 + 16 + r16) * ldb + kb;

  f32x8 c00 = {0.f,0.f,0.f,0.f,0.f,0.f,0.f,0.f};
  f32x8 c01 = c00, c10 = c00, c11 = c00;

  const int Kmain = Kreal & ~31;
  for (int k0 = 0; k0 < Kmain; k0 += 32) {
    bf16x16 a0, a1;
    if constexpr (AF32) {
      a0 = load_frag_f32<true>(a0f + k0);
      a1 = load_frag_f32<true>(a1f + k0);
    } else {
      a0 = load_frag_bf16(a0h + k0);
      a1 = load_frag_bf16(a1h + k0);
    }
    bf16x16 b0 = load_frag_bf16(b0p + k0);
    bf16x16 b1 = load_frag_bf16(b1p + k0);
    c00 = wmma_bf16(a0, b0, c00);
    c01 = wmma_bf16(a0, b1, c01);
    c10 = wmma_bf16(a1, b0, c10);
    c11 = wmma_bf16(a1, b1, c11);
  }
  if constexpr (AF32) {
    // K tail (Kreal % 32 == 16 in this pipeline): low 16 K valid, high half 0.
    if (Kreal & 31) {
      const int k0 = Kmain;
      bf16x16 a0 = load_frag_f32<false>(a0f + k0);
      bf16x16 a1 = load_frag_f32<false>(a1f + k0);
      bf16x16 b0 = load_frag_bf16(b0p + k0);   // B pad cols are zero
      bf16x16 b1 = load_frag_bf16(b1p + k0);
      c00 = wmma_bf16(a0, b0, c00);
      c01 = wmma_bf16(a0, b1, c01);
      c10 = wmma_bf16(a1, b0, c10);
      c11 = wmma_bf16(a1, b1, c11);
    }
  }

  // ---- epilogue ----
  // C layout: lane&15 = N, (lane>>4)*8 + vgpr = M.
  const int nl = lane & 15;
  const int mh = (lane >> 4) << 3;
#pragma unroll
  for (int t = 0; t < 4; ++t) {
    const f32x8& c = (t == 0) ? c00 : (t == 1) ? c01 : (t == 2) ? c10 : c11;
    const int gm = m0 + ((t >> 1) << 4);
    const int gn = n0 + ((t & 1) << 4) + nl;
    if (gn >= Nreal) continue;
    const float bv = bias ? bias[gn] : 0.0f;
#pragma unroll
    for (int i = 0; i < 8; ++i) {
      const int m = gm + mh + i;
      if (m >= M) continue;
      float v = c[i] + bv;
      if (relu && v < 0.0f) v = 0.0f;
      if (Cf) Cf[(size_t)m * ldcf + gn] = v;
      if (Cb) Cb[(size_t)m * ldcb + coffb + gn] = (bf16_t)v;
      if (Ct) Ct[(size_t)gn * ldct + m] = (bf16_t)v;
    }
  }
}

// ---------------------------------------------------------------------------
// Helper kernels
// ---------------------------------------------------------------------------
// f32 [rows x cols] (ld = lds) -> bf16 [rows x ldd] (pads stay zero)
__global__ void cvt2bf(const float* __restrict__ src, int rows, int cols, int lds,
                       bf16_t* __restrict__ dst, int ldd)
{
  long idx = (long)blockIdx.x * blockDim.x + threadIdx.x;
  int r = (int)(idx / cols), c = (int)(idx % cols);
  if (r >= rows) return;
  dst[(size_t)r * ldd + c] = (bf16_t)src[(size_t)r * lds + c];
}

// Wt[n][f] = W[n/H][f][n%H]   (W: [R,F,H] f32) -> bf16 [kNW x kKF]
__global__ void packW(const float* __restrict__ W, bf16_t* __restrict__ Wt)
{
  long idx = (long)blockIdx.x * blockDim.x + threadIdx.x;
  int n = (int)(idx / kF), f = (int)(idx % kF);
  if (n >= kR * kH) return;
  Wt[(size_t)n * kKF + f] =
      (bf16_t)W[((size_t)(n >> 6) * kF + f) * kH + (n & 63)];
}

// dst[n][k] = src[k][n]  (src: [K x N] f32 row-major) -> bf16 [Npad x ldd]
__global__ void packT(const float* __restrict__ src, int K, int N,
                      bf16_t* __restrict__ dst, int ldd)
{
  long idx = (long)blockIdx.x * blockDim.x + threadIdx.x;
  int n = (int)(idx / K), k = (int)(idx % K);
  if (n >= N) return;
  dst[(size_t)n * ldd + k] = (bf16_t)src[(size_t)k * N + n];
}

// Per-column mean / rsqrt(var+eps) over M rows.  grid.x == C, block 256.
__global__ void colstats(const float* __restrict__ X, int M, int ld,
                         float* __restrict__ stats)
{
  int c = blockIdx.x;
  float s = 0.f, s2 = 0.f;
  for (int m = threadIdx.x; m < M; m += 256) {
    float v = X[(size_t)m * ld + c];
    s += v; s2 += v * v;
  }
  __shared__ float sh0[256], sh1[256];
  sh0[threadIdx.x] = s; sh1[threadIdx.x] = s2;
  __syncthreads();
  for (int o = 128; o > 0; o >>= 1) {
    if ((int)threadIdx.x < o) {
      sh0[threadIdx.x] += sh0[threadIdx.x + o];
      sh1[threadIdx.x] += sh1[threadIdx.x + o];
    }
    __syncthreads();
  }
  if (threadIdx.x == 0) {
    float mu  = sh0[0] / (float)M;
    float var = sh1[0] / (float)M - mu * mu;
    stats[c]       = mu;
    stats[128 + c] = rsqrtf(var + 1e-5f);
  }
}

// y = relu((x - mu) * rstd * g + beta) -> bf16 at dst[m][coff + c];
// columns [C, Cpad) written as zero (keeps padded K clean for the next GEMM).
__global__ void bn_relu_store(const float* __restrict__ X, int M, int ldin,
                              int C, int Cpad,
                              const float* __restrict__ stats,
                              const float* __restrict__ g,
                              const float* __restrict__ beta,
                              bf16_t* __restrict__ dst, int ldd, int coff)
{
  long idx = (long)blockIdx.x * blockDim.x + threadIdx.x;
  int m = (int)(idx / Cpad), c = (int)(idx % Cpad);
  if (m >= M) return;
  float v = 0.f;
  if (c < C) {
    float x = X[(size_t)m * ldin + c];
    v = (x - stats[c]) * stats[128 + c] * g[c] + beta[c];
    v = v > 0.f ? v : 0.f;
  }
  dst[(size_t)m * ldd + coff + c] = (bf16_t)v;
}

// ---------------------------------------------------------------------------
// Launcher
// ---------------------------------------------------------------------------
extern "C" void kernel_launch(void* const* d_in, const int* in_sizes, int n_in,
                              void* d_out, int out_size, void* d_ws, size_t ws_size,
                              hipStream_t stream)
{
  (void)in_sizes; (void)n_in; (void)out_size; (void)ws_size;

  const float* feature_u   = (const float*)d_in[0];
  const float* feature_v   = (const float*)d_in[1];
  const float* M_u         = (const float*)d_in[2];
  const float* M_v         = (const float*)d_in[3];
  const float* W           = (const float*)d_in[4];
  const float* Q           = (const float*)d_in[5];
  const float* side_u      = (const float*)d_in[6];
  const float* side_v      = (const float*)d_in[7];
  const float* w_side_u    = (const float*)d_in[8];
  const float* b_side_u    = (const float*)d_in[9];
  const float* g_side_u    = (const float*)d_in[10];
  const float* beta_side_u = (const float*)d_in[11];
  const float* w_side_v    = (const float*)d_in[12];
  const float* b_side_v    = (const float*)d_in[13];
  const float* g_side_v    = (const float*)d_in[14];
  const float* beta_side_v = (const float*)d_in[15];
  const float* w_cat_u     = (const float*)d_in[16];
  const float* b_cat_u     = (const float*)d_in[17];
  const float* g_cat_u     = (const float*)d_in[18];
  const float* beta_cat_u  = (const float*)d_in[19];
  const float* w_cat_v     = (const float*)d_in[20];
  const float* b_cat_v     = (const float*)d_in[21];
  const float* g_cat_v     = (const float*)d_in[22];
  const float* beta_cat_v  = (const float*)d_in[23];

  float* score = (float*)d_out;

  // ---- workspace carve-up (256B aligned) ----
  char* ws = (char*)d_ws;
  size_t off = 0;
  auto carve = [&](size_t bytes) -> char* {
    char* p = ws + off;
    off = (off + bytes + 255) & ~(size_t)255;
    return p;
  };
  bf16_t* Wt_bf  = (bf16_t*)carve((size_t)kNW * kKF * 2);   // W_flat^T (bf16)
  bf16_t* Put    = (bf16_t*)carve((size_t)kNW * kKUV * 2);  // P_u^T [320pad x Upad]
  bf16_t* Pvt    = (bf16_t*)carve((size_t)kNW * kKUV * 2);  // P_v^T
  bf16_t* catu   = (bf16_t*)carve((size_t)kU * kCAT * 2);
  bf16_t* catv   = (bf16_t*)carve((size_t)kV * kCAT * 2);
  bf16_t* wsu_t  = (bf16_t*)carve((size_t)128 * kSF * 2);
  bf16_t* wsv_t  = (bf16_t*)carve((size_t)128 * kSF * 2);
  bf16_t* wcu_t  = (bf16_t*)carve((size_t)128 * kCAT * 2);
  bf16_t* wcv_t  = (bf16_t*)carve((size_t)128 * kCAT * 2);
  float*  pre_su = (float*)carve((size_t)kU * kSH * 4);
  float*  pre_sv = (float*)carve((size_t)kV * kSH * 4);
  float*  pre_cu = (float*)carve((size_t)kU * kLDO * 4);
  float*  pre_cv = (float*)carve((size_t)kV * kLDO * 4);
  bf16_t* embu   = (bf16_t*)carve((size_t)kU * kKO * 2);
  bf16_t* embv   = (bf16_t*)carve((size_t)kV * kKO * 2);
  bf16_t* Qp     = (bf16_t*)carve((size_t)kR * kNQ * kKO * 2);
  bf16_t* Gv     = (bf16_t*)carve((size_t)kR * kNGV * kKO * 2);
  float*  st_su  = (float*)carve(256 * 4);
  float*  st_sv  = (float*)carve(256 * 4);
  float*  st_cu  = (float*)carve(256 * 4);
  float*  st_cv  = (float*)carve(256 * 4);
  const size_t total = off;

  // Zero all pads (in-graph, deterministic; ~15 MB)
  hipMemsetAsync(d_ws, 0, total, stream);

  auto g1 = [](long n) { return dim3((unsigned)((n + 255) / 256)); };
  const dim3 blk(256);
  const int  myt64  = (kU + 63) / 64;     // 32 M-tiles for 64-row blocks
  const int  myt128 = (kU + 127) / 128;   // 16 M-tiles for 128-row blocks

  // ---- stage 0: weight packing (small) ----
  packW<<<g1((long)kR * kH * kF), blk, 0, stream>>>(W, Wt_bf);
  packT<<<g1((long)kSF * kSH), blk, 0, stream>>>(w_side_u, kSF, kSH, wsu_t, kSF);
  packT<<<g1((long)kSF * kSH), blk, 0, stream>>>(w_side_v, kSF, kSH, wsv_t, kSF);
  packT<<<g1((long)kCAT * kO), blk, 0, stream>>>(w_cat_u, kCAT, kO, wcu_t, kCAT);
  packT<<<g1((long)kCAT * kO), blk, 0, stream>>>(w_cat_v, kCAT, kO, wcv_t, kCAT);
  for (int r = 0; r < kR; ++r)
    cvt2bf<<<g1((long)kO * kO), blk, 0, stream>>>(
        Q + (size_t)r * kO * kO, kO, kO, kO, Qp + (size_t)r * kNQ * kKO, kKO);

  // ---- stage 1: P = feature @ W_flat  (A = f32 features, fused cast) ----
  // cat cols 320..639 (bf16) + transposed copy for stage 2.
  gemm_nt<4, true><<<dim3(kNW / 128, myt64), blk, 0, stream>>>(
      feature_v, Wt_bf, kV, kF, kF, kKF,
      catv, kCAT, kR * kH,   Pvt, kKUV,   nullptr, 0,
      nullptr, kR * kH, 0);
  gemm_nt<4, true><<<dim3(kNW / 128, myt64), blk, 0, stream>>>(
      feature_u, Wt_bf, kU, kF, kF, kKF,
      catu, kCAT, kR * kH,   Put, kKUV,   nullptr, 0,
      nullptr, kR * kH, 0);

  // ---- stage 2: hidden = relu(M[r] @ pre[r])  (A = f32 M slabs, read once) ----
  for (int r = 0; r < kR; ++r) {
    gemm_nt<2, true><<<dim3(1, myt128), blk, 0, stream>>>(
        M_u + (size_t)r * kU * kV, Pvt + (size_t)r * kH * kKUV, kU, kV, kV, kKUV,
        catu, kCAT, r * kH,   nullptr, 0,   nullptr, 0,
        nullptr, kH, 1);
    gemm_nt<2, true><<<dim3(1, myt128), blk, 0, stream>>>(
        M_v + (size_t)r * kV * kU, Put + (size_t)r * kH * kKUV, kV, kU, kU, kKUV,
        catv, kCAT, r * kH,   nullptr, 0,   nullptr, 0,
        nullptr, kH, 1);
  }

  // ---- stage 3: side linear (A = f32 side features) -> BN -> relu ----
  gemm_nt<2, true><<<dim3(1, myt128), blk, 0, stream>>>(
      side_u, wsu_t, kU, kSF, kSF, kSF,
      nullptr, 0, 0,   nullptr, 0,   pre_su, kSH,
      b_side_u, kSH, 0);
  gemm_nt<2, true><<<dim3(1, myt128), blk, 0, stream>>>(
      side_v, wsv_t, kV, kSF, kSF, kSF,
      nullptr, 0, 0,   nullptr, 0,   pre_sv, kSH,
      b_side_v, kSH, 0);
  colstats<<<kSH, blk, 0, stream>>>(pre_su, kU, kSH, st_su);
  colstats<<<kSH, blk, 0, stream>>>(pre_sv, kV, kSH, st_sv);
  bn_relu_store<<<g1((long)kU * kSH), blk, 0, stream>>>(
      pre_su, kU, kSH, kSH, kSH, st_su, g_side_u, beta_side_u, catu, kCAT, 2 * kR * kH);
  bn_relu_store<<<g1((long)kV * kSH), blk, 0, stream>>>(
      pre_sv, kV, kSH, kSH, kSH, st_sv, g_side_v, beta_side_v, catv, kCAT, 2 * kR * kH);

  // ---- stage 4: cat linear -> BN -> relu -> embed (bf16, K padded to 96) ----
  gemm_nt<2, false><<<dim3(kNQ / 64, myt128), blk, 0, stream>>>(
      catu, wcu_t, kU, kCAT, kCAT, kCAT,
      nullptr, 0, 0,   nullptr, 0,   pre_cu, kLDO,
      b_cat_u, kO, 0);
  gemm_nt<2, false><<<dim3(kNQ / 64, myt128), blk, 0, stream>>>(
      catv, wcv_t, kV, kCAT, kCAT, kCAT,
      nullptr, 0, 0,   nullptr, 0,   pre_cv, kLDO,
      b_cat_v, kO, 0);
  colstats<<<kO, blk, 0, stream>>>(pre_cu, kU, kLDO, st_cu);
  colstats<<<kO, blk, 0, stream>>>(pre_cv, kV, kLDO, st_cv);
  bn_relu_store<<<g1((long)kU * kKO), blk, 0, stream>>>(
      pre_cu, kU, kLDO, kO, kKO, st_cu, g_cat_u, beta_cat_u, embu, kKO, 0);
  bn_relu_store<<<g1((long)kV * kKO), blk, 0, stream>>>(
      pre_cv, kV, kLDO, kO, kKO, st_cv, g_cat_v, beta_cat_v, embv, kKO, 0);

  // ---- stage 5: G_v[r] = embed_v @ Q[r]^T  (bf16, zero-padded cols) ----
  for (int r = 0; r < kR; ++r) {
    gemm_nt<2, false><<<dim3(kNQ / 64, myt128), blk, 0, stream>>>(
        embv, Qp + (size_t)r * kNQ * kKO, kV, kKO, kKO, kKO,
        Gv + (size_t)r * kNGV * kKO, kKO, 0,   nullptr, 0,   nullptr, 0,
        nullptr, kKO, 0);
  }

  // ---- stage 6: score[r] = embed_u @ G_v[r]^T  -> f32 output ----
  for (int r = 0; r < kR; ++r) {
    gemm_nt<4, false><<<dim3(kNGV / 128, myt64), blk, 0, stream>>>(
        embu, Gv + (size_t)r * kNGV * kKO, kU, kKO, kKO, kKO,
        nullptr, 0, 0,   nullptr, 0,
        score + (size_t)r * kU * kV, kV,
        nullptr, kV, 0);
  }
}
